// CriticNetwork_79894981640578
// MI455X (gfx1250) — matrix operand order repaired and checked by reading, not compiled, not run on hardware
//
#include <hip/hip_runtime.h>
#include <hip/hip_bf16.h>

// ---------------------------------------------------------------------------
// CDNA5 (gfx1250) GCN critic network.
//   x[50000,128] -> GCNConv(relu) -> GCNConv(relu) -> mean -> MLP head -> [1]
// GEMMs via v_wmma_f32_16x16x32_f16 (f16 in, f32 acc); edge scatter via
// wave-per-edge hardware global_atomic_add_f32 with SGPR-uniform edge loads.
// ---------------------------------------------------------------------------

typedef __attribute__((ext_vector_type(16))) _Float16 v16h;
typedef __attribute__((ext_vector_type(8)))  _Float16 v8h;
typedef __attribute__((ext_vector_type(8)))  float    v8f;

union H16 { v16h v; v8h h[2]; };

#define FDIM 128

__device__ __forceinline__ void atomAddF32(float* addr, float val) {
  // Lowers to global_atomic_add_f32 (no CAS loop) on AMD.
  unsafeAtomicAdd(addr, val);
}

// ---------------- degree / normalization ----------------

__global__ void k_init_deg(float* __restrict__ deg, int n) {
  int i = blockIdx.x * blockDim.x + threadIdx.x;
  if (i < n) deg[i] = 1.0f;            // self-loop
}

__global__ void k_count_deg(const int* __restrict__ dst, float* __restrict__ deg,
                            int E) {
  int e = blockIdx.x * blockDim.x + threadIdx.x;
  if (e < E) atomAddF32(&deg[dst[e]], 1.0f);
}

__global__ void k_rsqrt(float* __restrict__ deg, int n) {
  int i = blockIdx.x * blockDim.x + threadIdx.x;
  if (i < n) deg[i] = rsqrtf(deg[i]);  // deg >= 1 always (self loop)
}

// ---------------- f32 -> f16 conversion ----------------

__global__ void k_cvt_f16(const float* __restrict__ src, _Float16* __restrict__ dst,
                          long long n) {
  long long i = (long long)blockIdx.x * blockDim.x + threadIdx.x;
  if (i < n) dst[i] = (_Float16)src[i];
}

// W[k,n] (row-major 128x128) -> Wt[n,k] f16 (so B-fragment K is contiguous)
__global__ void k_cvt_w_t(const float* __restrict__ W, _Float16* __restrict__ Wt) {
  int idx = blockIdx.x * blockDim.x + threadIdx.x;   // 16384 threads
  int n = idx & 127, k = idx >> 7;
  Wt[n * FDIM + k] = (_Float16)W[k * FDIM + n];
}

// ---------------- WMMA GEMM: C[M,128] = A[M,128] @ W[128,128] ----------------
// A: f16 row-major. Bt: f16 [n*128+k] = W[k][n]. One wave -> one 16x16 C tile.

__global__ __launch_bounds__(256) void k_gemm_wmma(
    const _Float16* __restrict__ A, const _Float16* __restrict__ Bt,
    float* __restrict__ C, int M) {
  int wave = threadIdx.x >> 5;
  int lane = threadIdx.x & 31;
  int gw   = blockIdx.x * 8 + wave;          // 8 waves/block = 8 N-tiles of one M-tile
  int tileN = gw & 7;
  int tileM = gw >> 3;
  if (tileM * 16 >= M) return;               // wave-uniform: EXEC stays all-1s
  int m0 = tileM * 16, n0 = tileN * 16;
  int lr  = lane & 15;
  int khi = lane >> 4;                       // 0: lanes 0-15, 1: lanes 16-31

  v8f acc = {};
  const _Float16* arow = A  + (long long)(m0 + lr) * FDIM;
  const _Float16* bcol = Bt + (long long)(n0 + lr) * FDIM;

#pragma unroll
  for (int k0 = 0; k0 < FDIM; k0 += 32) {
    // A 16x32 f16 layout: lanes<16 halves0-7 = K[k0..k0+7], halves8-15 = K[k0+16..k0+23]
    //                     lanes>=16 shift K by +8.
    H16 a, b;
    a.h[0] = *(const v8h*)(arow + k0 + khi * 8);
    a.h[1] = *(const v8h*)(arow + k0 + 16 + khi * 8);
    // B 32x16 f16 layout: same K order as A (halves0-7 = K[k0+8*khi..], halves8-15 = K[k0+16+8*khi..]) of col n0+lr
    b.h[0] = *(const v8h*)(bcol + k0 + khi * 8);
    b.h[1] = *(const v8h*)(bcol + k0 + 16 + khi * 8);
    acc = __builtin_amdgcn_wmma_f32_16x16x32_f16(
        /*neg_a=*/false, a.v, /*neg_b=*/false, b.v,
        /*c_mod=*/(short)0, acc, /*reuse_a=*/false, /*reuse_b=*/false);
  }

  // C layout: VGPR v -> row m0 + v + 8*khi, col n0 + lr
#pragma unroll
  for (int v = 0; v < 8; ++v) {
    C[(long long)(m0 + v + khi * 8) * FDIM + n0 + lr] = acc[v];
  }
}

// ---------------- GCN aggregation ----------------

// out[i,:] = lin[i,:] * dinv[i]^2   (self-loop term; also zero-initializes out)
__global__ void k_self_loop(const float* __restrict__ lin, const float* __restrict__ dinv,
                            float* __restrict__ out, int n) {
  long long idx = (long long)blockIdx.x * blockDim.x + threadIdx.x;
  if (idx < (long long)n * FDIM) {
    int i = (int)(idx >> 7);
    float w = dinv[i];
    out[idx] = lin[idx] * w * w;
  }
}

// out[dst,:] += lin[src,:] * dinv[src] * dinv[dst]
// One wave per edge: edge index + norm live in SGPRs (uniform loads), each lane
// gathers one float4 of lin[src] and issues 4 hardware f32 atomics.
__global__ __launch_bounds__(256) void k_edge_agg(
    const int* __restrict__ ei, const float* __restrict__ lin,
    const float* __restrict__ dinv, float* __restrict__ out, int E) {
  int lane = threadIdx.x & 31;
  int e = __builtin_amdgcn_readfirstlane(blockIdx.x * 8 + (threadIdx.x >> 5));
  if (e >= E) return;                       // wave-uniform
  int s = ei[e];
  int d = ei[E + e];
  float w = dinv[s] * dinv[d];
  float4 v = ((const float4*)(lin + (size_t)s * FDIM))[lane];
  float* o = out + (size_t)d * FDIM + lane * 4;
  atomAddF32(o + 0, v.x * w);
  atomAddF32(o + 1, v.y * w);
  atomAddF32(o + 2, v.z * w);
  atomAddF32(o + 3, v.w * w);
}

// layer-1 epilogue: h16 = (f16)relu(agg + b)   (f32 h1 is never materialized)
__global__ void k_bias_relu_cvt(const float* __restrict__ h, const float* __restrict__ b,
                                _Float16* __restrict__ h16, int n) {
  long long idx = (long long)blockIdx.x * blockDim.x + threadIdx.x;
  if (idx < (long long)n * FDIM) {
    int j = (int)(idx & 127);
    h16[idx] = (_Float16)fmaxf(h[idx] + b[j], 0.0f);
  }
}

// layer-2 epilogue fused with mean pool: mean[j] += sum_rows relu(agg + b) / n
__global__ void k_bias_relu_mean(const float* __restrict__ h, const float* __restrict__ b,
                                 float* __restrict__ mean, int n) {
  const int CH = 256;
  int j = threadIdx.x;                      // blockDim = 128
  float bj = b[j];
  long long r0 = (long long)blockIdx.x * CH;
  float s = 0.0f;
  for (int r = 0; r < CH; ++r) {
    long long row = r0 + r;
    if (row < n) s += fmaxf(h[row * FDIM + j] + bj, 0.0f);
  }
  atomAddF32(&mean[j], s * (1.0f / n));
}

__global__ void k_zero128(float* __restrict__ p) { p[threadIdx.x] = 0.0f; }

// ---------------- MLP head (single block, 128 threads) ----------------

__global__ void k_head(const float* __restrict__ mean,
                       const float* __restrict__ n1e, const float* __restrict__ n2e,
                       const float* __restrict__ aoh, const float* __restrict__ ap,
                       const float* __restrict__ Wp,  const float* __restrict__ bp,
                       const float* __restrict__ W1,  const float* __restrict__ b1,
                       const float* __restrict__ W2,  const float* __restrict__ b2,
                       float* __restrict__ out) {
  __shared__ float cat[388];
  __shared__ float hh[128];
  __shared__ float red[128];
  int j = threadIdx.x;  // 0..127

  float s1 = 0.0f, s2 = 0.0f;
  for (int k = 0; k < FDIM; ++k) {
    float w = Wp[k * FDIM + j];
    s1 += n1e[k] * w;
    s2 += n2e[k] * w;
  }
  cat[j]       = mean[j];
  cat[128 + j] = fmaxf(s1 + bp[j], 0.0f);
  cat[256 + j] = fmaxf(s2 + bp[j], 0.0f);
  if (j < 3)  cat[384 + j] = aoh[j];
  if (j == 0) cat[387]     = ap[0];
  __syncthreads();

  float a = 0.0f;
  for (int i = 0; i < 388; ++i) a += cat[i] * W1[i * FDIM + j];
  hh[j] = fmaxf(a + b1[j], 0.0f);
  __syncthreads();

  red[j] = hh[j] * W2[j];
  __syncthreads();
  for (int s = 64; s > 0; s >>= 1) {
    if (j < s) red[j] += red[j + s];
    __syncthreads();
  }
  if (j == 0) out[0] = red[0] + b2[0];
}

// ---------------------------------------------------------------------------

extern "C" void kernel_launch(void* const* d_in, const int* in_sizes, int n_in,
                              void* d_out, int out_size, void* d_ws, size_t ws_size,
                              hipStream_t stream) {
  const float* x   = (const float*)d_in[0];
  const int*   ei  = (const int*)  d_in[1];
  const float* aoh = (const float*)d_in[2];
  const float* n1e = (const float*)d_in[3];
  const float* n2e = (const float*)d_in[4];
  const float* ap  = (const float*)d_in[5];
  const float* Wg1 = (const float*)d_in[6];
  const float* bg1 = (const float*)d_in[7];
  const float* Wg2 = (const float*)d_in[8];
  const float* bg2 = (const float*)d_in[9];
  const float* Wp  = (const float*)d_in[10];
  const float* bp  = (const float*)d_in[11];
  const float* W1  = (const float*)d_in[12];
  const float* b1  = (const float*)d_in[13];
  const float* W2  = (const float*)d_in[14];
  const float* b2  = (const float*)d_in[15];

  const int N = in_sizes[0] / FDIM;        // 50000
  const int E = in_sizes[1] / 2;           // 1600000

  // Workspace layout (~64.3 MB)
  auto align256 = [](size_t o) { return (o + 255) & ~(size_t)255; };
  char* w = (char*)d_ws;
  size_t o = 0;
  float*    dinv = (float*)(w + o);    o = align256(o + (size_t)N * sizeof(float));
  _Float16* a16  = (_Float16*)(w + o); o = align256(o + (size_t)N * FDIM * 2);
  _Float16* w16  = (_Float16*)(w + o); o = align256(o + (size_t)FDIM * FDIM * 2);
  float*    lin  = (float*)(w + o);    o = align256(o + (size_t)N * FDIM * 4);
  float*    hbuf = (float*)(w + o);    o = align256(o + (size_t)N * FDIM * 4);
  float*    mnb  = (float*)(w + o);    o = align256(o + 128 * 4);

  const long long NF = (long long)N * FDIM;
  const int gNF = (int)((NF + 255) / 256);
  const int gN  = (N + 255) / 256;
  const int gE  = (E + 255) / 256;
  const int gEW = (E + 7) / 8;             // wave-per-edge, 8 waves/block
  const int gMM = (N + 15) / 16;           // one block per M-tile (8 waves = 8 N-tiles)

  // --- degree normalization: dinv = rsqrt(1 + indeg) ---
  k_init_deg <<<gN,  256, 0, stream>>>(dinv, N);
  k_count_deg<<<gE,  256, 0, stream>>>(ei + E, dinv, E);
  k_rsqrt    <<<gN,  256, 0, stream>>>(dinv, N);

  // --- layer 1 ---
  k_cvt_f16      <<<gNF, 256, 0, stream>>>(x, a16, NF);
  k_cvt_w_t      <<<64,  256, 0, stream>>>(Wg1, w16);
  k_gemm_wmma    <<<gMM, 256, 0, stream>>>(a16, w16, lin, N);
  k_self_loop    <<<gNF, 256, 0, stream>>>(lin, dinv, hbuf, N);
  k_edge_agg     <<<gEW, 256, 0, stream>>>(ei, lin, dinv, hbuf, E);
  k_bias_relu_cvt<<<gNF, 256, 0, stream>>>(hbuf, bg1, a16, N);

  // --- layer 2 ---
  k_cvt_w_t      <<<64,  256, 0, stream>>>(Wg2, w16);
  k_gemm_wmma    <<<gMM, 256, 0, stream>>>(a16, w16, lin, N);
  k_self_loop    <<<gNF, 256, 0, stream>>>(lin, dinv, hbuf, N);
  k_edge_agg     <<<gEW, 256, 0, stream>>>(ei, lin, dinv, hbuf, E);

  // --- fused bias+relu+mean pool, then head ---
  k_zero128       <<<1, 128, 0, stream>>>(mnb);
  k_bias_relu_mean<<<(N + 255) / 256, 128, 0, stream>>>(hbuf, bg2, mnb, N);
  k_head          <<<1, 128, 0, stream>>>(mnb, n1e, n2e, aoh, ap, Wp, bp, W1, b1, W2, b2,
                                          (float*)d_out);
}
